// ComplexNetAttentionBase_86569360818369
// MI455X (gfx1250) — 16-bit and 8-bit matrix operand order repaired and checked by reading, compiled, not run on hardware
//
#include <hip/hip_runtime.h>
#include <cstdint>
#include <cstddef>


typedef __attribute__((ext_vector_type(8)))  int      v8i;
typedef __attribute__((ext_vector_type(8)))  float    v8f;
typedef __attribute__((ext_vector_type(16))) _Float16 v16h;

#define TOK   2048
#define DIM   2048
#define SEQ   1024
#define HDIM  128

union AV16 { v16h h; int i[8]; };

__global__ void act_quant_kernel(const float* __restrict__ xr,
                                 const float* __restrict__ xi,
                                 int width,
                                 int8_t* __restrict__ qr,
                                 int8_t* __restrict__ qi,
                                 float* __restrict__ descale) {
  const int t = blockIdx.x;
  const float* r = xr + (size_t)t * width;
  const float* im = xi + (size_t)t * width;
  __shared__ float red[256];
  float amax = 0.f;
  for (int c = threadIdx.x; c < width; c += blockDim.x)
    amax = fmaxf(amax, fmaxf(fabsf(r[c]), fabsf(im[c])));
  red[threadIdx.x] = amax;
  __syncthreads();
  for (int o = 128; o > 0; o >>= 1) {
    if ((int)threadIdx.x < o) red[threadIdx.x] = fmaxf(red[threadIdx.x], red[threadIdx.x + o]);
    __syncthreads();
  }
  amax = fmaxf(red[0], 1e-5f);
  const float s = 127.0f / amax;
  if (threadIdx.x == 0) descale[t] = amax / 127.0f;
  int8_t* qrow_r = qr + (size_t)t * width;
  int8_t* qrow_i = qi + (size_t)t * width;
  for (int c = threadIdx.x; c < width; c += blockDim.x) {
    float vr = fminf(fmaxf(rintf(r[c]  * s), -128.f), 127.f);
    float vi = fminf(fmaxf(rintf(im[c] * s), -128.f), 127.f);
    qrow_r[c] = (int8_t)vr;
    qrow_i[c] = (int8_t)vi;
  }
}

__global__ void wmag_partial_kernel(const float* __restrict__ wr,
                                    const float* __restrict__ wi,
                                    size_t n, float* __restrict__ partial) {
  __shared__ float red[256];
  const size_t stride = (size_t)gridDim.x * blockDim.x;
  float v = 0.f;
  for (size_t k = (size_t)blockIdx.x * blockDim.x + threadIdx.x; k < n; k += stride)
    v += sqrtf(wr[k] * wr[k] + wi[k] * wi[k]);
  red[threadIdx.x] = v;
  __syncthreads();
  for (int o = 128; o > 0; o >>= 1) {
    if ((int)threadIdx.x < o) red[threadIdx.x] += red[threadIdx.x + o];
    __syncthreads();
  }
  if (threadIdx.x == 0) partial[blockIdx.x] = red[0];
}

__global__ void wmag_final_kernel(const float* __restrict__ partial,
                                  float inv_n, float* __restrict__ mag) {
  __shared__ float red[256];
  red[threadIdx.x] = partial[threadIdx.x];
  __syncthreads();
  for (int o = 128; o > 0; o >>= 1) {
    if ((int)threadIdx.x < o) red[threadIdx.x] += red[threadIdx.x + o];
    __syncthreads();
  }
  if (threadIdx.x == 0) mag[0] = red[0] * inv_n;
}

__global__ void wquant_pack_kernel(const float* __restrict__ wr,
                                   const float* __restrict__ wi,
                                   int K, int N,
                                   int8_t* __restrict__ pr,
                                   int8_t* __restrict__ pi) {
  const size_t idx = (size_t)blockIdx.x * blockDim.x + threadIdx.x;
  if (idx >= (size_t)K * N) return;
  const int k = (int)(idx / N), n = (int)(idx % N);
  const float r = wr[idx], im = wi[idx];
  const bool use_re = fabsf(r) >= fabsf(im);
  const int8_t qr = use_re ? (r  >= 0.f ? (int8_t)1 : (int8_t)-1) : (int8_t)0;
  const int8_t qi = use_re ? (int8_t)0 : (im >= 0.f ? (int8_t)1 : (int8_t)-1);
  const size_t p = ((size_t)(k >> 6) * N + n) * 64 + (k & 63);
  pr[p] = qr;
  pi[p] = qi;
}

struct KBlk { v8i ar, ai, br, bi; };

__device__ __forceinline__ void load_blk(KBlk& kb,
                                         const int8_t* __restrict__ arow_r,
                                         const int8_t* __restrict__ arow_i,
                                         const int8_t* __restrict__ bw_r,
                                         const int8_t* __restrict__ bw_i,
                                         int k0, int h16, int N, int ncol) {
  const int2* par = (const int2*)(arow_r + k0 + h16 * 16);
  const int2* pai = (const int2*)(arow_i + k0 + h16 * 16);
  int2 a0 = par[0], a1 = par[1], a2 = par[4], a3 = par[5];
  int2 c0 = pai[0], c1 = pai[1], c2 = pai[4], c3 = pai[5];
  kb.ar[0] = a0.x; kb.ar[1] = a0.y; kb.ar[2] = a1.x; kb.ar[3] = a1.y;
  kb.ar[4] = a2.x; kb.ar[5] = a2.y; kb.ar[6] = a3.x; kb.ar[7] = a3.y;
  kb.ai[0] = c0.x; kb.ai[1] = c0.y; kb.ai[2] = c1.x; kb.ai[3] = c1.y;
  kb.ai[4] = c2.x; kb.ai[5] = c2.y; kb.ai[6] = c3.x; kb.ai[7] = c3.y;
  const size_t rec = ((size_t)(k0 >> 6) * N + ncol) * 64;
  const int4* pbr = (const int4*)(bw_r + rec);
  const int4* pbi = (const int4*)(bw_i + rec);
  int4 b0 = pbr[h16], b1 = pbr[2 + h16];
  int4 d0 = pbi[h16], d1 = pbi[2 + h16];
  kb.br[0] = b0.x; kb.br[1] = b0.y; kb.br[2] = b0.z; kb.br[3] = b0.w;
  kb.br[4] = b1.x; kb.br[5] = b1.y; kb.br[6] = b1.z; kb.br[7] = b1.w;
  kb.bi[0] = d0.x; kb.bi[1] = d0.y; kb.bi[2] = d0.z; kb.bi[3] = d0.w;
  kb.bi[4] = d1.x; kb.bi[5] = d1.y; kb.bi[6] = d1.z; kb.bi[7] = d1.w;
}

__device__ __forceinline__ void wmma4(const KBlk& kb, v8i& rr, v8i& ii, v8i& ri, v8i& ir) {
  rr = __builtin_amdgcn_wmma_i32_16x16x64_iu8(true, kb.ar, true, kb.br, rr, false, false);
  ii = __builtin_amdgcn_wmma_i32_16x16x64_iu8(true, kb.ai, true, kb.bi, ii, false, false);
  ri = __builtin_amdgcn_wmma_i32_16x16x64_iu8(true, kb.ar, true, kb.bi, ri, false, false);
  ir = __builtin_amdgcn_wmma_i32_16x16x64_iu8(true, kb.ai, true, kb.br, ir, false, false);
}

template <int MODE>
__global__ void __launch_bounds__(32)
cgemm_i8_kernel(const int8_t* __restrict__ ar, const int8_t* __restrict__ ai,
                const int8_t* __restrict__ br, const int8_t* __restrict__ bi,
                const float* __restrict__ descale, const float* __restrict__ magp,
                int N, int nheads,
                _Float16* __restrict__ h_re, _Float16* __restrict__ h_im,
                float* __restrict__ f_re, float* __restrict__ f_im) {
  const int lane = threadIdx.x;
  const int l16 = lane & 15;
  const int h16 = lane >> 4;
  const int tiles_n = N >> 4;
  const int tm = blockIdx.x / tiles_n;
  const int tn = blockIdx.x % tiles_n;
  const int ncol = tn * 16 + l16;

  const int8_t* arow_r = ar + (size_t)(tm * 16 + l16) * DIM;
  const int8_t* arow_i = ai + (size_t)(tm * 16 + l16) * DIM;

  v8i accrr, accii, accri, accir;
#pragma unroll
  for (int e = 0; e < 8; ++e) { accrr[e] = 0; accii[e] = 0; accri[e] = 0; accir[e] = 0; }

  KBlk cur, nxt;
  load_blk(cur, arow_r, arow_i, br, bi, 0, h16, N, ncol);
  for (int k0 = 0; k0 < DIM; k0 += 128) {
    __builtin_prefetch(arow_r + k0 + 256);
    load_blk(nxt, arow_r, arow_i, br, bi, k0 + 64, h16, N, ncol);
    wmma4(cur, accrr, accii, accri, accir);
    if (k0 + 128 < DIM)
      load_blk(cur, arow_r, arow_i, br, bi, k0 + 128, h16, N, ncol);
    wmma4(nxt, accrr, accii, accri, accir);
  }

  const float mag = magp[0];
#pragma unroll
  for (int r = 0; r < 8; ++r) {
    const int M = r + 8 * h16;
    const int t = tm * 16 + M;
    const float scv = descale[t] * mag;
    const float ore = scv * (float)(accrr[r] - accii[r]);
    const float oim = scv * (float)(accri[r] + accir[r]);
    if (MODE == 2) {
      f_re[(size_t)t * N + ncol] = ore;
      f_im[(size_t)t * N + ncol] = oim;
    } else {
      const int b = t >> 10, s = t & (SEQ - 1);
      const int head = ncol >> 7, d = ncol & (HDIM - 1);
      size_t o;
      if (MODE == 0) o = (((size_t)(b * nheads + head)) * SEQ + s) * HDIM + d;
      else           o = (((size_t)(b * nheads + head)) * HDIM + d) * SEQ + s;
      h_re[o] = (_Float16)ore;
      h_im[o] = (_Float16)oim;
    }
  }
}

__global__ void rope_kernel(_Float16* __restrict__ x, size_t total) {
  const size_t idx = (size_t)blockIdx.x * blockDim.x + threadIdx.x;
  if (idx >= total) return;
  const int d = (int)(idx & 63);
  const size_t rest = idx >> 6;
  const int s = (int)(rest & (SEQ - 1));
  const size_t bh = rest >> 10;
  const size_t base = (bh * SEQ + s) * HDIM;
  const float inv_freq = __powf(10000.0f, -(float)(2 * d) / 128.0f);
  const float fr = (float)s * inv_freq;
  const float c = __cosf(fr), sn = __sinf(fr);
  const float x0 = (float)x[base + d];
  const float x1 = (float)x[base + d + 64];
  x[base + d]      = (_Float16)(x0 * c - x1 * sn);
  x[base + d + 64] = (_Float16)(x1 * c + x0 * sn);
}

__device__ __forceinline__ void async_ld16(unsigned lds_off, const void* gaddr) {
  asm volatile("global_load_async_to_lds_b128 %0, %1, off"
               :: "v"(lds_off), "v"((unsigned long long)(uintptr_t)gaddr)
               : "memory");
}

__global__ void __launch_bounds__(32)
attn_kernel(const _Float16* __restrict__ q_re, const _Float16* __restrict__ q_im,
            const _Float16* __restrict__ k_re, const _Float16* __restrict__ k_im,
            const _Float16* __restrict__ vt_re, const _Float16* __restrict__ vt_im,
            float* __restrict__ o_re, float* __restrict__ o_im) {
  const int lane = threadIdx.x;
  const int l16 = lane & 15;
  const int h16 = lane >> 4;
  const int bid = blockIdx.x;
  const int qt = bid & 63;
  const int h  = (bid >> 6) & 15;
  const int b  = bid >> 10;
  const int kv = h >> 2;
  const int qbase = qt * 16;

  const _Float16* qr_base = q_re  + ((size_t)(b * 16 + h)  * SEQ) * HDIM;
  const _Float16* qi_base = q_im  + ((size_t)(b * 16 + h)  * SEQ) * HDIM;
  const _Float16* kr_base = k_re  + ((size_t)(b * 4  + kv) * SEQ) * HDIM;
  const _Float16* ki_base = k_im  + ((size_t)(b * 4  + kv) * SEQ) * HDIM;
  const _Float16* vr_base = vt_re + ((size_t)(b * 4  + kv) * HDIM) * SEQ;
  const _Float16* vi_base = vt_im + ((size_t)(b * 4  + kv) * HDIM) * SEQ;

  __shared__ _Float16 Kl[2 * 2 * 32 * 128];
  __shared__ _Float16 Pl[16 * 32];
  const unsigned kl_base = (unsigned)(uintptr_t)(void*)&Kl[0];

  const int nchunks = qt / 2 + 1;

  auto issue_chunk = [&](int kc, int buf) {
    const int key = kc * 32 + lane;
    const char* gr = (const char*)(kr_base + (size_t)key * HDIM);
    const char* gi = (const char*)(ki_base + (size_t)key * HDIM);
    const unsigned lr = kl_base + ((unsigned)(buf * 2 + 0) * 32 + lane) * 256;
    const unsigned li = kl_base + ((unsigned)(buf * 2 + 1) * 32 + lane) * 256;
#pragma unroll
    for (int j = 0; j < 16; ++j) {
      async_ld16(lr + j * 16, gr + j * 16);
      async_ld16(li + j * 16, gi + j * 16);
    }
  };

  issue_chunk(0, 0);

  AV16 qa_re[4], qa_im[4];
  const _Float16* qrow_r = qr_base + (size_t)(qbase + l16) * HDIM;
  const _Float16* qrow_i = qi_base + (size_t)(qbase + l16) * HDIM;
#pragma unroll
  for (int ch = 0; ch < 4; ++ch) {
#pragma unroll
    for (int j = 0; j < 8; ++j) {
      const int K = ch * 32 + (j < 4 ? 2 * j : 16 + 2 * (j - 4)) + h16 * 8;
      qa_re[ch].i[j] = *(const int*)(qrow_r + K);
      qa_im[ch].i[j] = *(const int*)(qrow_i + K);
    }
  }

  v8f o_r[8], o_i[8];
#pragma unroll
  for (int dt = 0; dt < 8; ++dt)
#pragma unroll
    for (int e = 0; e < 8; ++e) { o_r[dt][e] = 0.f; o_i[dt][e] = 0.f; }

  float m_s[8], l_s[8];
#pragma unroll
  for (int r = 0; r < 8; ++r) { m_s[r] = -1e30f; l_s[r] = 0.f; }

  const float scale = 0.08838834764831845f;

  for (int kc = 0; kc < nchunks; ++kc) {
    const int kb = kc * 32;
    const int buf = kc & 1;

    if (kc + 1 < nchunks) {
      issue_chunk(kc + 1, (kc + 1) & 1);
      asm volatile("s_wait_asynccnt 0x20" ::: "memory");
    } else {
      asm volatile("s_wait_asynccnt 0x0" ::: "memory");
    }

    v8f sc[2];
#pragma unroll
    for (int g = 0; g < 2; ++g)
#pragma unroll
      for (int e = 0; e < 8; ++e) sc[g][e] = 0.f;

#pragma unroll
    for (int g = 0; g < 2; ++g) {
      const _Float16* klr = &Kl[((size_t)(buf * 2 + 0) * 32 + g * 16 + l16) * 128];
      const _Float16* kli = &Kl[((size_t)(buf * 2 + 1) * 32 + g * 16 + l16) * 128];
      AV16 kt_r[4], kt_i[4];
#pragma unroll
      for (int ch = 0; ch < 4; ++ch)
#pragma unroll
        for (int j = 0; j < 8; ++j) {
          const int K = ch * 32 + (j < 4 ? 2 * j : 16 + 2 * (j - 4)) + h16 * 8;
          kt_r[ch].i[j] = *(const int*)(klr + K);
          kt_i[ch].i[j] = *(const int*)(kli + K);
        }
#pragma unroll
      for (int ch = 0; ch < 4; ++ch) {
        sc[g] = __builtin_amdgcn_wmma_f32_16x16x32_f16(false, qa_re[ch].h, false, kt_r[ch].h,
                                                       (short)0, sc[g], false, false);
        sc[g] = __builtin_amdgcn_wmma_f32_16x16x32_f16(false, qa_im[ch].h, false, kt_i[ch].h,
                                                       (short)0, sc[g], false, false);
      }
    }

    float p0[8], p1[8];
#pragma unroll
    for (int r = 0; r < 8; ++r) {
      const int M = r + 8 * h16;
      const int qrow = qbase + M;
      const int key0 = kb + l16, key1 = kb + 16 + l16;
      float s0 = sc[0][r] * scale; if (key0 > qrow) s0 = -1e30f;
      float s1 = sc[1][r] * scale; if (key1 > qrow) s1 = -1e30f;
      float mx = fmaxf(s0, s1);
#pragma unroll
      for (int msk = 1; msk < 16; msk <<= 1) mx = fmaxf(mx, __shfl_xor(mx, msk, 32));
      const float m_new = fmaxf(m_s[r], mx);
      const float corr = __expf(m_s[r] - m_new);
      const float e0 = __expf(s0 - m_new);
      const float e1 = __expf(s1 - m_new);
      float rs = e0 + e1;
#pragma unroll
      for (int msk = 1; msk < 16; msk <<= 1) rs += __shfl_xor(rs, msk, 32);
      l_s[r] = l_s[r] * corr + rs;
      m_s[r] = m_new;
      p0[r] = e0; p1[r] = e1;
#pragma unroll
      for (int dt = 0; dt < 8; ++dt) { o_r[dt][r] *= corr; o_i[dt][r] *= corr; }
    }

    __syncthreads();
#pragma unroll
    for (int r = 0; r < 8; ++r) {
      const int M = r + 8 * h16;
      Pl[M * 32 + l16]      = (_Float16)p0[r];
      Pl[M * 32 + 16 + l16] = (_Float16)p1[r];
    }
    __syncthreads();
    AV16 pa;
#pragma unroll
    for (int j = 0; j < 8; ++j) {
      const int K = h16 * 16 + 2 * j;
      pa.i[j] = *(const int*)&Pl[l16 * 32 + K];
    }

#pragma unroll
    for (int dt = 0; dt < 8; ++dt) {
      const _Float16* vr_row = vr_base + (size_t)(dt * 16 + l16) * SEQ + kb + h16 * 16;
      const _Float16* vi_row = vi_base + (size_t)(dt * 16 + l16) * SEQ + kb + h16 * 16;
      AV16 vb_r, vb_i;
#pragma unroll
      for (int j = 0; j < 8; ++j) {
        vb_r.i[j] = *(const int*)(vr_row + 2 * j);
        vb_i.i[j] = *(const int*)(vi_row + 2 * j);
      }
      o_r[dt] = __builtin_amdgcn_wmma_f32_16x16x32_f16(false, pa.h, false, vb_r.h,
                                                       (short)0, o_r[dt], false, false);
      o_i[dt] = __builtin_amdgcn_wmma_f32_16x16x32_f16(false, pa.h, false, vb_i.h,
                                                       (short)0, o_i[dt], false, false);
    }
  }

#pragma unroll
  for (int r = 0; r < 8; ++r) {
    const int M = r + 8 * h16;
    const int t = b * SEQ + qbase + M;
    const float inv_l = 1.0f / l_s[r];
#pragma unroll
    for (int dt = 0; dt < 8; ++dt) {
      const int col = h * HDIM + dt * 16 + l16;
      o_re[(size_t)t * DIM + col] = o_r[dt][r] * inv_l;
      o_im[(size_t)t * DIM + col] = o_i[dt][r] * inv_l;
    }
  }
}

extern "C" void kernel_launch(void* const* d_in, const int* in_sizes, int n_in,
                              void* d_out, int out_size, void* d_ws, size_t ws_size,
                              hipStream_t stream) {
  (void)in_sizes; (void)n_in; (void)out_size; (void)ws_size;

  const float* hr    = (const float*)d_in[0];
  const float* hi    = (const float*)d_in[1];
  const float* wq_re = (const float*)d_in[2];
  const float* wq_im = (const float*)d_in[3];
  const float* wk_re = (const float*)d_in[4];
  const float* wk_im = (const float*)d_in[5];
  const float* wv_re = (const float*)d_in[6];
  const float* wv_im = (const float*)d_in[7];
  const float* wo_re = (const float*)d_in[8];
  const float* wo_im = (const float*)d_in[9];

  const int NQ = 2048, NKV = 512;

  char* ws = (char*)d_ws;
  size_t off = 0;
  auto alloc = [&](size_t bytes) -> char* {
    char* p = ws + off;
    off += (bytes + 255) & ~(size_t)255;
    return p;
  };

  int8_t* a8_r    = (int8_t*)alloc((size_t)TOK * DIM);
  int8_t* a8_i    = (int8_t*)alloc((size_t)TOK * DIM);
  float*  a_scale = (float*) alloc((size_t)TOK * 4);
  int8_t* wq8_r   = (int8_t*)alloc((size_t)DIM * NQ);
  int8_t* wq8_i   = (int8_t*)alloc((size_t)DIM * NQ);
  int8_t* wk8_r   = (int8_t*)alloc((size_t)DIM * NKV);
  int8_t* wk8_i   = (int8_t*)alloc((size_t)DIM * NKV);
  int8_t* wv8_r   = (int8_t*)alloc((size_t)DIM * NKV);
  int8_t* wv8_i   = (int8_t*)alloc((size_t)DIM * NKV);
  int8_t* wo8_r   = (int8_t*)alloc((size_t)DIM * DIM);
  int8_t* wo8_i   = (int8_t*)alloc((size_t)DIM * DIM);
  float*  part    = (float*) alloc(4 * 256 * sizeof(float));
  float*  mags    = (float*) alloc(4 * sizeof(float));
  _Float16* q16_re = (_Float16*)alloc((size_t)2 * 16 * SEQ * HDIM * 2);
  _Float16* q16_im = (_Float16*)alloc((size_t)2 * 16 * SEQ * HDIM * 2);
  _Float16* k16_re = (_Float16*)alloc((size_t)2 * 4  * SEQ * HDIM * 2);
  _Float16* k16_im = (_Float16*)alloc((size_t)2 * 4  * SEQ * HDIM * 2);
  _Float16* v16_re = (_Float16*)alloc((size_t)2 * 4  * SEQ * HDIM * 2);
  _Float16* v16_im = (_Float16*)alloc((size_t)2 * 4  * SEQ * HDIM * 2);
  float* obuf_re  = (float*) alloc((size_t)TOK * DIM * 4);
  float* obuf_im  = (float*) alloc((size_t)TOK * DIM * 4);
  int8_t* o8_r    = (int8_t*)alloc((size_t)TOK * DIM);
  int8_t* o8_i    = (int8_t*)alloc((size_t)TOK * DIM);
  float*  o_scale = (float*) alloc((size_t)TOK * 4);

  act_quant_kernel<<<TOK, 256, 0, stream>>>(hr, hi, DIM, a8_r, a8_i, a_scale);

  wmag_partial_kernel<<<256, 256, 0, stream>>>(wq_re, wq_im, (size_t)DIM * NQ,  part + 0);
  wmag_partial_kernel<<<256, 256, 0, stream>>>(wk_re, wk_im, (size_t)DIM * NKV, part + 256);
  wmag_partial_kernel<<<256, 256, 0, stream>>>(wv_re, wv_im, (size_t)DIM * NKV, part + 512);
  wmag_partial_kernel<<<256, 256, 0, stream>>>(wo_re, wo_im, (size_t)DIM * DIM, part + 768);
  wmag_final_kernel<<<1, 256, 0, stream>>>(part + 0,   1.0f / ((float)DIM * NQ),  mags + 0);
  wmag_final_kernel<<<1, 256, 0, stream>>>(part + 256, 1.0f / ((float)DIM * NKV), mags + 1);
  wmag_final_kernel<<<1, 256, 0, stream>>>(part + 512, 1.0f / ((float)DIM * NKV), mags + 2);
  wmag_final_kernel<<<1, 256, 0, stream>>>(part + 768, 1.0f / ((float)DIM * DIM), mags + 3);

  wquant_pack_kernel<<<(DIM * NQ  + 255) / 256, 256, 0, stream>>>(wq_re, wq_im, DIM, NQ,  wq8_r, wq8_i);
  wquant_pack_kernel<<<(DIM * NKV + 255) / 256, 256, 0, stream>>>(wk_re, wk_im, DIM, NKV, wk8_r, wk8_i);
  wquant_pack_kernel<<<(DIM * NKV + 255) / 256, 256, 0, stream>>>(wv_re, wv_im, DIM, NKV, wv8_r, wv8_i);
  wquant_pack_kernel<<<(DIM * DIM + 255) / 256, 256, 0, stream>>>(wo_re, wo_im, DIM, DIM, wo8_r, wo8_i);

  cgemm_i8_kernel<0><<<(TOK / 16) * (NQ  / 16), 32, 0, stream>>>(
      a8_r, a8_i, wq8_r, wq8_i, a_scale, mags + 0, NQ, 16,
      q16_re, q16_im, (float*)nullptr, (float*)nullptr);
  cgemm_i8_kernel<0><<<(TOK / 16) * (NKV / 16), 32, 0, stream>>>(
      a8_r, a8_i, wk8_r, wk8_i, a_scale, mags + 1, NKV, 4,
      k16_re, k16_im, (float*)nullptr, (float*)nullptr);
  cgemm_i8_kernel<1><<<(TOK / 16) * (NKV / 16), 32, 0, stream>>>(
      a8_r, a8_i, wv8_r, wv8_i, a_scale, mags + 2, NKV, 4,
      v16_re, v16_im, (float*)nullptr, (float*)nullptr);

  {
    const size_t qtot = (size_t)2 * 16 * SEQ * 64;
    const size_t ktot = (size_t)2 * 4  * SEQ * 64;
    rope_kernel<<<(unsigned)((qtot + 255) / 256), 256, 0, stream>>>(q16_re, qtot);
    rope_kernel<<<(unsigned)((ktot + 255) / 256), 256, 0, stream>>>(k16_re, ktot);
  }

  attn_kernel<<<2 * 16 * (SEQ / 16), 32, 0, stream>>>(
      q16_re, q16_im, k16_re, k16_im, v16_re, v16_im, obuf_re, obuf_im);

  act_quant_kernel<<<TOK, 256, 0, stream>>>(obuf_re, obuf_im, DIM, o8_r, o8_i, o_scale);

  float* out_re = (float*)d_out;
  float* out_im = out_re + (size_t)TOK * DIM;
  cgemm_i8_kernel<2><<<(TOK / 16) * (DIM / 16), 32, 0, stream>>>(
      o8_r, o8_i, wo8_r, wo8_i, o_scale, mags + 3, DIM, 0,
      (_Float16*)nullptr, (_Float16*)nullptr, out_re, out_im);
}
